// BDH_GPU_25752623907041
// MI455X (gfx1250) — matrix operand order repaired and checked by reading, not compiled, not run on hardware
//
#include <hip/hip_runtime.h>

typedef __attribute__((ext_vector_type(16))) __bf16 v16bf;
typedef __attribute__((ext_vector_type(8)))  __bf16 v8bf;
typedef __attribute__((ext_vector_type(8)))  float  v8f;

#define DEV __device__ __forceinline__

DEV __bf16 f2bf(float f) {
  unsigned u = __builtin_bit_cast(unsigned, f);
  unsigned r = (u >> 16) & 1u;
  u += 0x7fffu + r;                       // round-to-nearest-even
  unsigned short h = (unsigned short)(u >> 16);
  return __builtin_bit_cast(__bf16, h);
}
DEV float bf2f(__bf16 b) {
  unsigned short h = __builtin_bit_cast(unsigned short, b);
  unsigned u = ((unsigned)h) << 16;
  return __builtin_bit_cast(float, u);
}
// assemble a 16-element bf16 fragment from two contiguous 16B LDS runs
DEV v16bf frag16(const __bf16* lo, const __bf16* hi) {
  v8bf l = *(const v8bf*)lo;
  v8bf h = *(const v8bf*)hi;
  v16bf r;
#pragma unroll
  for (int e = 0; e < 8; ++e) { r[e] = l[e]; r[e + 8] = h[e]; }
  return r;
}
DEV v8f wmma_bf16(v16bf a, v16bf b, v8f c) {
  return __builtin_amdgcn_wmma_f32_16x16x32_bf16(false, a, false, b, (short)0, c,
                                                 false, false);
}
// one 32-k step of a 32x32 per-wave tile: 4 frag loads -> 4 WMMAs
DEV void gemm_step(const __bf16 (*As)[40], const __bf16 (*Bs)[40],
                   int wy, int wx, int l16, int half,
                   v8f& c00, v8f& c01, v8f& c10, v8f& c11) {
  const __bf16* a0p = &As[wy * 32 + l16][half * 8];
  const __bf16* a1p = &As[wy * 32 + 16 + l16][half * 8];
  v16bf a0 = frag16(a0p, a0p + 16);
  v16bf a1 = frag16(a1p, a1p + 16);
  const __bf16* b0p = &Bs[wx * 32 + l16][half * 8];
  const __bf16* b1p = &Bs[wx * 32 + 16 + l16][half * 8];
  v16bf b0 = frag16(b0p, b0p + 16);
  v16bf b1 = frag16(b1p, b1p + 16);
  c00 = wmma_bf16(a0, b0, c00);
  c01 = wmma_bf16(a0, b1, c01);
  c10 = wmma_bf16(a1, b0, c10);
  c11 = wmma_bf16(a1, b1, c11);
}
DEV float block_sum(float v, float* red) {
  int t = threadIdx.x;
  red[t] = v; __syncthreads();
  for (int o = blockDim.x >> 1; o > 0; o >>= 1) {
    if (t < o) red[t] += red[t + o];
    __syncthreads();
  }
  float r = red[0]; __syncthreads();
  return r;
}
DEV float block_max(float v, float* red) {
  int t = threadIdx.x;
  red[t] = v; __syncthreads();
  for (int o = blockDim.x >> 1; o > 0; o >>= 1) {
    if (t < o) red[t] = fmaxf(red[t], red[t + o]);
    __syncthreads();
  }
  float r = red[0]; __syncthreads();
  return r;
}

// ------------------------------------------------ decoder (H,D,n) -> bf16 Wt[(h*n+j)*D + d]
__global__ void cvt_dec_t_k(const float* __restrict__ in, __bf16* __restrict__ out,
                            int H, int D, int n) {
  long long total = (long long)H * D * n;
  long long i = (long long)blockIdx.x * blockDim.x + threadIdx.x;
  long long st = (long long)gridDim.x * blockDim.x;
  for (; i < total; i += st) {
    long long c = i / D;               // output row (h*n + j)
    int d = (int)(i - c * D);
    long long h = c / n, j = c - h * n;
    out[i] = f2bf(in[(h * D + d) * (long long)n + j]);
  }
}
// ------------------------------------------------ generic transpose f32(R,C) -> bf16 out[c*R + r]
__global__ void cvt_t_k(const float* __restrict__ in, __bf16* __restrict__ out,
                        long long R, long long C) {
  long long total = R * C;
  long long i = (long long)blockIdx.x * blockDim.x + threadIdx.x;
  long long st = (long long)gridDim.x * blockDim.x;
  for (; i < total; i += st) {
    long long c = i / R, r = i - c * R;
    out[i] = f2bf(in[r * C + c]);
  }
}

// ------------------------------------------------ embedding gather + LN (-> v f32, vb bf16, vbT bf16)
__global__ void embed_ln_k(const int* __restrict__ idx, const float* __restrict__ wte,
                           float* __restrict__ v, __bf16* __restrict__ vb,
                           __bf16* __restrict__ vbT, int D, int T) {
  __shared__ float red[256];
  int row = blockIdx.x, d = threadIdx.x;
  long long tok = idx[row];
  float x = wte[tok * (long long)D + d];
  float mean = block_sum(x, red) / (float)D;
  float xc = x - mean;
  float var = block_sum(xc * xc, red) / (float)D;
  float y = xc * rsqrtf(var + 1e-5f);
  v[(long long)row * D + d] = y;
  __bf16 yb = f2bf(y);
  vb[(long long)row * D + d] = yb;
  int b = row / T, t = row - b * T;
  vbT[((long long)b * D + d) * T + t] = yb;
}

// ------------------------------------------------ row LN (fp32 -> bf16)
__global__ void ln_rows_k(const float* __restrict__ in, __bf16* __restrict__ outb, int D) {
  __shared__ float red[256];
  long long i = (long long)blockIdx.x * D + threadIdx.x;
  float x = in[i];
  float mean = block_sum(x, red) / (float)D;
  float xc = x - mean;
  float var = block_sum(xc * xc, red) / (float)D;
  outb[i] = f2bf(xc * rsqrtf(var + 1e-5f));
}

// ------------------------------------------------ v = ln(v + ln(Z))  (-> v, vb, vbT)
__global__ void add_ln_ln_k(const float* __restrict__ Z, float* __restrict__ v,
                            __bf16* __restrict__ vb, __bf16* __restrict__ vbT,
                            int D, int T) {
  __shared__ float red[256];
  int row = blockIdx.x, d = threadIdx.x;
  long long i = (long long)row * D + d;
  float z = Z[i];
  float m1 = block_sum(z, red) / (float)D;
  float zc = z - m1;
  float v1 = block_sum(zc * zc, red) / (float)D;
  float u = zc * rsqrtf(v1 + 1e-5f);
  float w = v[i] + u;
  float m2 = block_sum(w, red) / (float)D;
  float wc = w - m2;
  float v2 = block_sum(wc * wc, red) / (float)D;
  float y = wc * rsqrtf(v2 + 1e-5f);
  v[i] = y;
  __bf16 yb = f2bf(y);
  vb[i] = yb;
  int b = row / T, t = row - b * T;
  vbT[((long long)b * D + d) * T + t] = yb;
}

// ------------------------------------------------ RoPE, vectorized 8 bf16 (=4 pairs) per thread
__global__ void rope_k(const __bf16* __restrict__ X, __bf16* __restrict__ Xr,
                       int T, int H, int n, long long totalVec) {
  long long iv = (long long)blockIdx.x * blockDim.x + threadIdx.x;
  long long st = (long long)gridDim.x * blockDim.x;
  const float negln = -9.210340371976184f;   // -ln(10000)
  const float invn = 1.0f / (float)n;
  for (; iv < totalVec; iv += st) {
    long long base = iv * 8;
    int inHead = (int)(base % n);
    long long r = base / n;                  // = ((b*T)+t)*H + h
    long long bt = r / H;
    int t = (int)(bt % T);
    v8bf xv = *(const v8bf*)(X + base);
    v8bf o;
#pragma unroll
    for (int q = 0; q < 4; ++q) {
      int p = (inHead >> 1) + q;             // pair index in head
      float inv = __expf(negln * ((float)(2 * p) * invn));
      float ang = (float)t * inv;
      float sn, cs;
      __sincosf(ang, &sn, &cs);
      float xe = bf2f(xv[2 * q]), xo = bf2f(xv[2 * q + 1]);
      o[2 * q]     = f2bf(xe * cs - xo * sn);
      o[2 * q + 1] = f2bf(xo * cs + xe * sn);
    }
    *(v8bf*)(Xr + base) = o;
  }
}

// ------------------------------------------------ bf16 WMMA GEMM, block tile 64Mx64N, 4 waves,
// per-wave 32x32 (4 accumulators), double-buffered LDS (one barrier per 32-k step).
// A row-major (M x K); Bt is B stored N-major: Bt[c*ldbT + k]. K must be a multiple of 64.
// z axis: batching (aBatch/bBatch/cBatch) and/or split-K (kStrideZ = slice K offset).
// mode 0: Cf = acc    mode 1: Cb = bf16(relu(acc))    mode 2: Cb = bf16(relu(acc)*Xmul)
__global__ void gemm64_k(const __bf16* __restrict__ A, long long lda, long long aBatch,
                         const __bf16* __restrict__ Bt, long long ldbT, long long bBatch,
                         float* __restrict__ Cf, __bf16* __restrict__ Cb,
                         const __bf16* __restrict__ Xmul,
                         long long ldc, long long cBatch,
                         int K, long long kStrideZ, int mode) {
  __shared__ __bf16 As[2][64][40];   // padded rows: bank-conflict-free 16B frag loads
  __shared__ __bf16 Bs[2][64][40];
  const int tid = threadIdx.x;
  const int wave = tid >> 5, lane = tid & 31;
  const int half = lane >> 4, l16 = lane & 15;
  const int wy = wave >> 1, wx = wave & 1;
  const long long c0 = (long long)blockIdx.x * 64;
  const long long m0 = (long long)blockIdx.y * 64;
  const long long z = blockIdx.z;
  const long long kOff = z * kStrideZ;
  const int ar = tid >> 2;              // 0..31 (also handles row ar+32)
  const int kq = (tid & 3) * 8;         // 0,8,16,24
  const __bf16* aptr = A + z * aBatch + (m0 + ar) * lda + kOff + kq;
  const __bf16* bptr = Bt + z * bBatch + (c0 + ar) * ldbT + kOff + kq;
  const long long a32 = 32 * lda, b32 = 32 * ldbT;
  v8f c00 = {}, c01 = {}, c10 = {}, c11 = {};
  // preload chunk 0 into buffer 0
  v8bf av0 = *(const v8bf*)aptr, av1 = *(const v8bf*)(aptr + a32);
  v8bf bv0 = *(const v8bf*)bptr, bv1 = *(const v8bf*)(bptr + b32);
  aptr += 32; bptr += 32;
  *(v8bf*)&As[0][ar][kq] = av0; *(v8bf*)&As[0][ar + 32][kq] = av1;
  *(v8bf*)&Bs[0][ar][kq] = bv0; *(v8bf*)&Bs[0][ar + 32][kq] = bv1;
  __syncthreads();
  for (int k0 = 0; k0 < K; k0 += 64) {
    // even step: compute buf0, fetch chunk (k0+32) -> buf1
    av0 = *(const v8bf*)aptr; av1 = *(const v8bf*)(aptr + a32);
    bv0 = *(const v8bf*)bptr; bv1 = *(const v8bf*)(bptr + b32);
    aptr += 32; bptr += 32;
    __builtin_prefetch(aptr, 0, 1);     // global_prefetch_b8 next chunk
    __builtin_prefetch(bptr, 0, 1);
    gemm_step(As[0], Bs[0], wy, wx, l16, half, c00, c01, c10, c11);
    *(v8bf*)&As[1][ar][kq] = av0; *(v8bf*)&As[1][ar + 32][kq] = av1;
    *(v8bf*)&Bs[1][ar][kq] = bv0; *(v8bf*)&Bs[1][ar + 32][kq] = bv1;
    __syncthreads();
    // odd step: compute buf1, fetch chunk (k0+64) -> buf0 (guard final step)
    const bool more = (k0 + 64) < K;
    if (more) {
      av0 = *(const v8bf*)aptr; av1 = *(const v8bf*)(aptr + a32);
      bv0 = *(const v8bf*)bptr; bv1 = *(const v8bf*)(bptr + b32);
      aptr += 32; bptr += 32;
    }
    gemm_step(As[1], Bs[1], wy, wx, l16, half, c00, c01, c10, c11);
    if (more) {
      *(v8bf*)&As[0][ar][kq] = av0; *(v8bf*)&As[0][ar + 32][kq] = av1;
      *(v8bf*)&Bs[0][ar][kq] = bv0; *(v8bf*)&Bs[0][ar + 32][kq] = bv1;
    }
    __syncthreads();
  }
  const long long ccB = c0 + wx * 32 + l16;
#pragma unroll
  for (int mt = 0; mt < 2; ++mt) {
#pragma unroll
    for (int nt = 0; nt < 2; ++nt) {
      v8f acc = (mt == 0) ? (nt == 0 ? c00 : c01) : (nt == 0 ? c10 : c11);
      const long long cc = ccB + nt * 16;
#pragma unroll
      for (int r = 0; r < 8; ++r) {
        long long m = m0 + wy * 32 + mt * 16 + r + half * 8;
        long long ci = z * cBatch + m * ldc + cc;
        if (mode == 0) {
          Cf[ci] = acc[r];
        } else {
          float vv = fmaxf(acc[r], 0.0f);
          if (mode == 2) vv *= bf2f(Xmul[ci]);
          Cb[ci] = f2bf(vv);
        }
      }
    }
  }
}

// ------------------------------------------------ causal QK^T scores (per b,h), 16Tx64S tile,
// double-buffered LDS; K-loop over n (multiple of 64).
__global__ void scores_k(const __bf16* __restrict__ Xr, float* __restrict__ S,
                         int T, int n, long long ldx, float scale, int H) {
  const int z = blockIdx.z;           // b*H + h
  const int b = z / H, h = z - b * H;
  const int t0 = blockIdx.y * 16;
  const int s0 = blockIdx.x * 64;
  if (s0 > t0 + 15) return;           // whole tile above diagonal
  __shared__ __bf16 Qs[2][16][40];
  __shared__ __bf16 Ks[2][64][40];
  const int tid = threadIdx.x;
  const int wave = tid >> 5, lane = tid & 31;
  const int half = lane >> 4, l16 = lane & 15;
  const int ar = tid >> 2;
  const int kq = (tid & 3) * 8;
  const __bf16* base = Xr + (long long)b * T * ldx + (long long)h * n;
  const __bf16* qptr = base + (long long)(t0 + (ar & 15)) * ldx + kq;  // tid<64 only
  const __bf16* kptr = base + (long long)(s0 + ar) * ldx + kq;
  const long long k32 = 32 * ldx;
  v8f acc = {};
  // preload chunk 0 into buffer 0
  v8bf qv = {};
  if (tid < 64) qv = *(const v8bf*)qptr;
  v8bf kv0 = *(const v8bf*)kptr, kv1 = *(const v8bf*)(kptr + k32);
  qptr += 32; kptr += 32;
  if (tid < 64) *(v8bf*)&Qs[0][ar & 15][kq] = qv;
  *(v8bf*)&Ks[0][ar][kq] = kv0; *(v8bf*)&Ks[0][ar + 32][kq] = kv1;
  __syncthreads();
  for (int k0 = 0; k0 < n; k0 += 64) {
    // even step: compute buf0, fetch (k0+32) -> buf1
    if (tid < 64) qv = *(const v8bf*)qptr;
    kv0 = *(const v8bf*)kptr; kv1 = *(const v8bf*)(kptr + k32);
    qptr += 32; kptr += 32;
    __builtin_prefetch(kptr, 0, 1);
    {
      v16bf af = frag16(&Qs[0][l16][half * 8], &Qs[0][l16][16 + half * 8]);
      const __bf16* brow = &Ks[0][wave * 16 + l16][half * 8];
      acc = wmma_bf16(af, frag16(brow, brow + 16), acc);
    }
    if (tid < 64) *(v8bf*)&Qs[1][ar & 15][kq] = qv;
    *(v8bf*)&Ks[1][ar][kq] = kv0; *(v8bf*)&Ks[1][ar + 32][kq] = kv1;
    __syncthreads();
    // odd step: compute buf1, fetch (k0+64) -> buf0 (guard final step)
    const bool more = (k0 + 64) < n;
    if (more) {
      if (tid < 64) qv = *(const v8bf*)qptr;
      kv0 = *(const v8bf*)kptr; kv1 = *(const v8bf*)(kptr + k32);
      qptr += 32; kptr += 32;
    }
    {
      v16bf af = frag16(&Qs[1][l16][half * 8], &Qs[1][l16][16 + half * 8]);
      const __bf16* brow = &Ks[1][wave * 16 + l16][half * 8];
      acc = wmma_bf16(af, frag16(brow, brow + 16), acc);
    }
    if (more) {
      if (tid < 64) *(v8bf*)&Qs[0][ar & 15][kq] = qv;
      *(v8bf*)&Ks[0][ar][kq] = kv0; *(v8bf*)&Ks[0][ar + 32][kq] = kv1;
    }
    __syncthreads();
  }
#pragma unroll
  for (int r = 0; r < 8; ++r) {
    int t = t0 + r + half * 8;
    int s = s0 + wave * 16 + l16;
    if (s <= t) S[(((long long)z * T) + t) * T + s] = acc[r] * scale;
  }
}

// ------------------------------------------------ causal softmax + head mean -> bf16
__global__ void softmax_mean_k(const float* __restrict__ S, __bf16* __restrict__ Pm,
                               int T, int H) {
  __shared__ float red[256];
  const int t = blockIdx.x, b = blockIdx.y, s = threadIdx.x;
  float pm = 0.0f;
  for (int hh = 0; hh < H; ++hh) {
    const float* row = S + (((long long)b * H + hh) * T + t) * (long long)T;
    float val = (s <= t) ? row[s] : -3.4e38f;
    float mx = block_max(val, red);
    float e = (s <= t) ? __expf(val - mx) : 0.0f;
    float sum = block_sum(e, red);
    pm += e / sum;
  }
  Pm[((long long)b * T + t) * T + s] = f2bf(pm / (float)H);
}

// ------------------------------------------------ deterministic split-K reduction
__global__ void reduce_slices_k(const float* __restrict__ part, float* __restrict__ out,
                                long long per, int slices) {
  long long i = (long long)blockIdx.x * blockDim.x + threadIdx.x;
  long long st = (long long)gridDim.x * blockDim.x;
  for (; i < per; i += st) {
    float s = 0.0f;
    for (int j = 0; j < slices; ++j) s += part[(long long)j * per + i];
    out[i] = s;
  }
}

// ================================================================ host side
extern "C" void kernel_launch(void* const* d_in, const int* in_sizes, int n_in,
                              void* d_out, int out_size, void* d_ws, size_t ws_size,
                              hipStream_t stream) {
  (void)in_sizes; (void)n_in; (void)out_size; (void)ws_size;
  const int B = 2, T = 256, D = 256, H = 4, VOC = 32000;
  const int n = 8192, N = H * n;            // 32768
  const int L = 6, KS = 16;                 // split-K slices for the encoder GEMM

  const int*   idx  = (const int*)d_in[0];
  const float* wte  = (const float*)d_in[1];
  const float* enc  = (const float*)d_in[2];
  const float* decx = (const float*)d_in[3];
  const float* decy = (const float*)d_in[4];
  const float* rdo  = (const float*)d_in[5];
  float* out = (float*)d_out;

  size_t off = 0;
  auto alloc = [&](size_t bytes) -> void* {
    void* p = (char*)d_ws + off;
    off += (bytes + 255) & ~(size_t)255;
    return p;
  };
  const long long nWx = (long long)H * D * n;
  const long long nEnc = (long long)N * D;
  const long long nRd = (long long)D * VOC;
  __bf16* WxT  = (__bf16*)alloc(nWx * 2);      // (N x D)
  __bf16* WyT  = (__bf16*)alloc(nWx * 2);      // (N x D)
  __bf16* EncT = (__bf16*)alloc(nEnc * 2);     // (D x N)
  __bf16* RdT  = (__bf16*)alloc(nRd * 2);      // (VOC x D)
  float*  v    = (float*)alloc((size_t)B * T * D * 4);
  __bf16* vb   = (__bf16*)alloc((size_t)B * T * D * 2);
  __bf16* vbT  = (__bf16*)alloc((size_t)B * D * T * 2);
  __bf16* X    = (__bf16*)alloc((size_t)B * T * N * 2);
  __bf16* Xr   = (__bf16*)alloc((size_t)B * T * N * 2);  // reused as Y
  float*  S    = (float*)alloc((size_t)B * H * T * T * 4);
  __bf16* Pm   = (__bf16*)alloc((size_t)B * T * T * 2);
  float*  a    = (float*)alloc((size_t)B * T * D * 4);
  __bf16* ab   = (__bf16*)alloc((size_t)B * T * D * 2);
  float*  Z    = (float*)alloc((size_t)B * T * D * 4);
  float*  Zp   = (float*)alloc((size_t)KS * B * T * D * 4);
  __bf16* Yb   = Xr;

  const float scale = 0.011048543456039806f;   // 1/sqrt(8192)
  const long long MBT = (long long)B * T;      // 512 rows

  // one-time bf16 weight transposes (~67MB bf16 -> resident in 192MB L2 thereafter)
  cvt_dec_t_k<<<4096, 256, 0, stream>>>(decx, WxT, H, D, n);
  cvt_dec_t_k<<<4096, 256, 0, stream>>>(decy, WyT, H, D, n);
  cvt_t_k<<<4096, 256, 0, stream>>>(enc, EncT, (long long)N, (long long)D);
  cvt_t_k<<<4096, 256, 0, stream>>>(rdo, RdT, (long long)D, (long long)VOC);

  embed_ln_k<<<B * T, 256, 0, stream>>>(idx, wte, v, vb, vbT, D, T);

  for (int l = 0; l < L; ++l) {
    // X = relu(v @ decoder_x) -> (B*T, N) bf16
    gemm64_k<<<dim3(N / 64, MBT / 64, 1), 128, 0, stream>>>(
        vb, D, 0, WxT, D, 0, nullptr, X, nullptr, N, 0, D, 0, 1);
    // RoPE
    rope_k<<<4096, 256, 0, stream>>>(X, Xr, T, H, n, (long long)B * T * N / 8);
    // causal scores per (b,h)
    scores_k<<<dim3(T / 64, T / 16, B * H), 128, 0, stream>>>(Xr, S, T, n, N, scale, H);
    // softmax + head mean
    softmax_mean_k<<<dim3(T, B), 256, 0, stream>>>(S, Pm, T, H);
    // a = attn_mean @ v  (batched over B; B operand = vbT, N-major)
    gemm64_k<<<dim3(D / 64, T / 64, B), 128, 0, stream>>>(
        Pm, T, (long long)T * T, vbT, T, (long long)D * T,
        a, nullptr, nullptr, D, (long long)T * D, T, 0, 0);
    // ab = ln(a)
    ln_rows_k<<<B * T, 256, 0, stream>>>(a, ab, D);
    // Y = relu(ab @ decoder_y) * X -> (B*T, N) bf16 (overwrites Xr)
    gemm64_k<<<dim3(N / 64, MBT / 64, 1), 128, 0, stream>>>(
        ab, D, 0, WyT, D, 0, nullptr, Yb, X, N, 0, D, 0, 2);
    // Z = Y @ encoder, K=32768: deterministic split-K, 16 slices of 2048
    gemm64_k<<<dim3(D / 64, MBT / 64, KS), 128, 0, stream>>>(
        Yb, N, 0, EncT, N, 0, Zp, nullptr, nullptr,
        D, MBT * D, N / KS, (long long)(N / KS), 0);
    reduce_slices_k<<<512, 256, 0, stream>>>(Zp, Z, MBT * D, KS);
    // v = ln(v + ln(Z))
    add_ln_ln_k<<<B * T, 256, 0, stream>>>(Z, v, vb, vbT, D, T);
  }

  // out = v @ readout -> (B*T, 32000) f32
  gemm64_k<<<dim3(VOC / 64, MBT / 64, 1), 128, 0, stream>>>(
      vb, D, 0, RdT, D, 0, out, nullptr, nullptr, VOC, 0, D, 0, 0);
}
